// DepthAwareGATv2_15522011808332
// MI455X (gfx1250) — matrix operand order repaired and checked by reading, not compiled, not run on hardware
//
#include <hip/hip_runtime.h>
#include <math.h>

typedef _Float16 half_t;
typedef __attribute__((ext_vector_type(16))) _Float16 v16h;
typedef __attribute__((ext_vector_type(8)))  float    v8f;

// ---------------- helpers ----------------
__device__ __forceinline__ float warp_sum32(float v) {
#pragma unroll
  for (int m = 16; m > 0; m >>= 1) v += __shfl_xor(v, m, 32);
  return v;
}

// float atomic max via signed/unsigned int ordering trick (valid for all finite floats)
__device__ __forceinline__ void atomic_max_float(float* addr, float val) {
  if (val >= 0.0f) atomicMax((int*)addr, __float_as_int(val));
  else             atomicMin((unsigned int*)addr, __float_as_uint(val));
}

// ---------------- tiny precompute: M[h][k] = sum_c lin_edge[h*32+c, k] * att_edge[h,c] ----------------
__global__ void k_edge_att_mats(const float* __restrict__ le1, const float* __restrict__ aw1,
                                const float* __restrict__ le2, const float* __restrict__ aw2,
                                float* __restrict__ M /* 32 floats: M1 then M2 */) {
  int t = threadIdx.x;
  if (t >= 32) return;
  int which = t >> 4;
  int h = (t >> 2) & 3;
  int k = t & 3;
  const float* le = which ? le2 : le1;
  const float* aw = which ? aw2 : aw1;
  float s = 0.f;
  for (int c = 0; c < 32; ++c) s += le[(h * 32 + c) * 4 + k] * aw[h * 32 + c];
  M[t] = s;
}

// ---------------- f32 -> f16 weight conversion for WMMA ----------------
__global__ void k_whalf(const float* __restrict__ c2, const float* __restrict__ jkw,
                        half_t* __restrict__ c2h, half_t* __restrict__ jkh) {
  int i = blockIdx.x * blockDim.x + threadIdx.x;
  if (i < 16384) c2h[i] = (half_t)c2[i];
  if (i < 49152) jkh[i] = (half_t)jkw[i];
}

// ---------------- edge encoder: relu(ea@W1^T+b1)@W2^T+b2 -> e[4]; a_edge = e @ M^T ----------------
__global__ __launch_bounds__(128)
void k_edge_encoder(const float* __restrict__ eattr,
                    const float* __restrict__ w1, const float* __restrict__ b1,
                    const float* __restrict__ w2, const float* __restrict__ b2,
                    const float* __restrict__ M,
                    float* __restrict__ ae1, float* __restrict__ ae2, int E) {
  __shared__ float sw1[256], sb1[32], sw2[128], sb2[4], sM[32];
  for (int i = threadIdx.x; i < 256; i += 128) sw1[i] = w1[i];
  if (threadIdx.x < 128) sw2[threadIdx.x] = w2[threadIdx.x];
  if (threadIdx.x < 32) { sb1[threadIdx.x] = b1[threadIdx.x]; sM[threadIdx.x] = M[threadIdx.x]; }
  if (threadIdx.x < 4) sb2[threadIdx.x] = b2[threadIdx.x];
  __syncthreads();
  int e = blockIdx.x * 128 + threadIdx.x;
  if (e >= E) return;
  float ea[8];
#pragma unroll
  for (int k = 0; k < 8; ++k) ea[k] = eattr[(size_t)e * 8 + k];
  float e4[4];
#pragma unroll
  for (int h = 0; h < 4; ++h) e4[h] = sb2[h];
  for (int c = 0; c < 32; ++c) {
    float t = sb1[c];
#pragma unroll
    for (int k = 0; k < 8; ++k) t += ea[k] * sw1[c * 8 + k];
    t = fmaxf(t, 0.f);
#pragma unroll
    for (int h = 0; h < 4; ++h) e4[h] += t * sw2[h * 32 + c];
  }
  float o1[4], o2[4];
#pragma unroll
  for (int h = 0; h < 4; ++h) {
    float s1 = 0.f, s2 = 0.f;
#pragma unroll
    for (int k = 0; k < 4; ++k) { s1 += e4[k] * sM[h * 4 + k]; s2 += e4[k] * sM[16 + h * 4 + k]; }
    o1[h] = s1; o2[h] = s2;
  }
  ((float4*)ae1)[e] = make_float4(o1[0], o1[1], o1[2], o1[3]);
  ((float4*)ae2)[e] = make_float4(o2[0], o2[1], o2[2], o2[3]);
}

// ---------------- input projections (K=16, exact f32): h0 = x@proj^T+b ; xs1 = x@c1_lin^T ----------------
__global__ __launch_bounds__(128)
void k_node_in(const float* __restrict__ x,
               const float* __restrict__ pw, const float* __restrict__ pb,
               const float* __restrict__ lw,
               float* __restrict__ h0, float* __restrict__ xs, int N) {
  __shared__ float xr[16];
  int n = blockIdx.x;
  if (threadIdx.x < 16) xr[threadIdx.x] = x[(size_t)n * 16 + threadIdx.x];
  __syncthreads();
  int d = threadIdx.x;
  float a = pb[d], b = 0.f;
#pragma unroll
  for (int k = 0; k < 16; ++k) {
    float xv = xr[k];
    a += xv * pw[d * 16 + k];
    b += xv * lw[d * 16 + k];
  }
  h0[(size_t)n * 128 + d] = a;
  xs[(size_t)n * 128 + d] = b;
}

// ---------------- per-node attention coefficients ----------------
__global__ void k_att(const float* __restrict__ xs,
                      const float* __restrict__ asw, const float* __restrict__ adw,
                      float* __restrict__ as_, float* __restrict__ ad_, int N) {
  int t = blockIdx.x * blockDim.x + threadIdx.x;
  if (t >= N * 4) return;
  int n = t >> 2, h = t & 3;
  float s = 0.f, d = 0.f;
  const float* row = xs + (size_t)n * 128 + h * 32;
#pragma unroll
  for (int c = 0; c < 32; ++c) { float v = row[c]; s += v * asw[h * 32 + c]; d += v * adw[h * 32 + c]; }
  as_[t] = s; ad_[t] = d;
}

// ---------------- init per-conv scratch ----------------
__global__ void k_init(float* __restrict__ amax, float* __restrict__ den,
                       float* __restrict__ agg, int N) {
  int i = blockIdx.x * blockDim.x + threadIdx.x;
  if (i < N * 128) agg[i] = 0.f;
  if (i < N * 4) { amax[i] = -3.0e38f; den[i] = 0.f; }
}

// ---------------- per-edge alpha + segment max ----------------
__global__ void k_edge_alpha(const int* __restrict__ src, const int* __restrict__ dst,
                             const float* __restrict__ as_, const float* __restrict__ ad_,
                             const float* __restrict__ ae, float* __restrict__ exb,
                             float* __restrict__ amax, int E) {
  int e = blockIdx.x * blockDim.x + threadIdx.x;
  if (e >= E) return;
  int s = src[e], d = dst[e];
  float4 a = ((const float4*)as_)[s];
  float4 b = ((const float4*)ad_)[d];
  float4 c = ((const float4*)ae)[e];
  float al[4] = {a.x + b.x + c.x, a.y + b.y + c.y, a.z + b.z + c.z, a.w + b.w + c.w};
#pragma unroll
  for (int h = 0; h < 4; ++h) {
    float v = al[h];
    v = v > 0.f ? v : 0.2f * v;   // leaky_relu(0.2)
    al[h] = v;
    atomic_max_float(&amax[d * 4 + h], v);
  }
  ((float4*)exb)[e] = make_float4(al[0], al[1], al[2], al[3]);
}

// ---------------- exp + segment sum ----------------
__global__ void k_edge_exp(const int* __restrict__ dst, float* __restrict__ exb,
                           const float* __restrict__ amax, float* __restrict__ den, int E) {
  int e = blockIdx.x * blockDim.x + threadIdx.x;
  if (e >= E) return;
  int d = dst[e];
  float4 al = ((const float4*)exb)[e];
  float4 m  = ((const float4*)amax)[d];
  float ex[4] = {expf(al.x - m.x), expf(al.y - m.y), expf(al.z - m.z), expf(al.w - m.w)};
#pragma unroll
  for (int h = 0; h < 4; ++h) atomicAdd(&den[d * 4 + h], ex[h]);
  ((float4*)exb)[e] = make_float4(ex[0], ex[1], ex[2], ex[3]);
}

// ---------------- weighted message scatter: agg[dst] += xs[src] * alpha ----------------
__global__ __launch_bounds__(128)
void k_message(const int* __restrict__ src, const int* __restrict__ dst,
               const float* __restrict__ exb, const float* __restrict__ den,
               const float* __restrict__ xs, float* __restrict__ agg, int E) {
  int t = threadIdx.x;
  int h = t >> 5;
  int base = blockIdx.x * 16;
#pragma unroll 1
  for (int i = 0; i < 16; ++i) {
    int e = base + i;
    if (e >= E) break;
    int s = src[e], d = dst[e];
    if (i + 1 < 16 && e + 1 < E) {
      int sn = src[e + 1];
      __builtin_prefetch(&xs[(size_t)sn * 128 + t], 0, 0);  // global_prefetch_b8
    }
    float w = exb[e * 4 + h] / (den[d * 4 + h] + 1e-16f);
    float m = xs[(size_t)s * 128 + t] * w;
    atomicAdd(&agg[(size_t)d * 128 + t], m);
  }
}

// ---------------- elu + residual + LayerNorm; emit f32 row + f16 row(s) into concat buffer ----------------
__global__ __launch_bounds__(256)
void k_post(const float* __restrict__ agg, const float* __restrict__ bias,
            const float* __restrict__ res, const float* __restrict__ g,
            const float* __restrict__ bb, float* __restrict__ hout,
            half_t* __restrict__ cat, int catcol, int rescol, int N) {
  int wave = threadIdx.x >> 5, lane = threadIdx.x & 31;
  int n = blockIdx.x * 8 + wave;
  if (n >= N) return;
  float v[4], rsv[4];
  float sum = 0.f, sq = 0.f;
#pragma unroll
  for (int i = 0; i < 4; ++i) {
    int c = i * 32 + lane;
    float a = agg[(size_t)n * 128 + c] + bias[c];
    a = a > 0.f ? a : expm1f(a);                 // elu
    float r = res[(size_t)n * 128 + c];
    rsv[i] = r;
    a += r;
    v[i] = a; sum += a; sq += a * a;
  }
  sum = warp_sum32(sum); sq = warp_sum32(sq);
  float mu = sum * (1.f / 128.f);
  float var = sq * (1.f / 128.f) - mu * mu;
  float rstd = rsqrtf(var + 1e-5f);
#pragma unroll
  for (int i = 0; i < 4; ++i) {
    int c = i * 32 + lane;
    float o = (v[i] - mu) * rstd * g[c] + bb[c];
    if (hout) hout[(size_t)n * 128 + c] = o;
    cat[(size_t)n * 384 + catcol + c] = (half_t)o;
    if (rescol >= 0) cat[(size_t)n * 384 + rescol + c] = (half_t)rsv[i];
  }
}

// ---------------- WMMA GEMM: out[M,128] = A[M,K](f16,lda) @ W[128,K](f16,ldw)^T (+bias) ----------------
// Block = 256 threads = 8 waves; block computes 16 rows x 128 cols; wave w -> 16x16 tile (cols w*16..).
// A tile (16 x K f16) is staged ONCE per block into LDS via GLOBAL_LOAD_ASYNC_TO_LDS_B128
// (ASYNCcnt-tracked, no VGPR round trip), so the 8 waves don't re-fetch it 8x from global.
// B (weights) stays on the global path: it is shared by all blocks and L2-resident.
template <int K>
__global__ __launch_bounds__(256)
void k_gemm_wmma(const half_t* __restrict__ A, int lda,
                 const half_t* __restrict__ W, int ldw,
                 float* __restrict__ out, const float* __restrict__ bias) {
  constexpr int KP = K + 8;                  // pad row stride by 8 halves to spread LDS banks
  __shared__ half_t sA[16 * KP];
  const int mbase = blockIdx.x * 16;

  // ---- async-copy A tile to LDS in 16B chunks ----
  constexpr int CPR = K / 8;                 // 16-byte chunks per row
  constexpr int TOTAL = 16 * CPR;
#pragma unroll
  for (int i = threadIdx.x; i < TOTAL; i += 256) {
    const int row = i / CPR;
    const int c = i % CPR;
    const half_t* gp = A + (size_t)(mbase + row) * lda + c * 8;
    unsigned lds = (unsigned)(size_t)&sA[row * KP + c * 8];  // LDS byte offset = addr[31:0]
    asm volatile("global_load_async_to_lds_b128 %0, %1, off"
                 :: "v"(lds), "v"(gp) : "memory");
  }
  asm volatile("s_wait_asynccnt 0x0" ::: "memory");  // this wave's async loads landed
  __syncthreads();                                   // all waves' chunks visible

  const int wave = threadIdx.x >> 5;
  const int lane = threadIdx.x & 31;
  const int hsel = lane >> 4;      // which K-half this lane holds
  const int l16  = lane & 15;
  const int nbase = wave * 16;
  const half_t* Ar   = &sA[l16 * KP];
  const half_t* Wcol = W + (size_t)(nbase + l16) * ldw;
  v8f acc = {};
#pragma unroll
  for (int kb = 0; kb < K; kb += 32) {
    v16h a, b;
#pragma unroll
    for (int j = 0; j < 4; ++j) {
      int k0 = kb + 2 * j + hsel * 8;        // A VGPR j   : K = 2j{,+1} (+8 for lanes 16-31)
      a[2 * j]     = Ar[k0];
      a[2 * j + 1] = Ar[k0 + 1];
      int k1 = kb + 16 + 2 * j + hsel * 8;   // A VGPR j+4 : K = 16+2j{,+1} (+8 for lanes 16-31)
      a[8 + 2 * j]     = Ar[k1];
      a[8 + 2 * j + 1] = Ar[k1 + 1];
    }
#pragma unroll
    for (int v = 0; v < 8; ++v) {
      int kv = kb + 2 * (v & 3) + (v >> 2) * 16 + hsel * 8;  // B VGPR v: K = 2(v&3){,+1} (+16 for v>=4) (+8 for lanes 16-31)
      b[2 * v]     = Wcol[kv];
      b[2 * v + 1] = Wcol[kv + 1];
    }
    acc = __builtin_amdgcn_wmma_f32_16x16x32_f16(false, a, false, b, (short)0, acc, false, false);
  }
#pragma unroll
  for (int r = 0; r < 8; ++r) {
    int row = mbase + r + hsel * 8;          // C/D: VGPR r -> M=r (lanes 0-15) / r+8 (lanes 16-31)
    int col = nbase + l16;
    float v = acc[r];
    if (bias) v += bias[col];
    out[(size_t)row * 128 + col] = v;
  }
}

// ---------------- classifier + log_softmax ----------------
__global__ __launch_bounds__(256)
void k_classifier(const float* __restrict__ hjk, const float* __restrict__ w,
                  const float* __restrict__ b, float* __restrict__ out, int N) {
  __shared__ float sw[640], sb[8];
  for (int i = threadIdx.x; i < 640; i += 256) sw[i] = w[i];
  if (threadIdx.x < 5) sb[threadIdx.x] = b[threadIdx.x];
  __syncthreads();
  int n = blockIdx.x * 256 + threadIdx.x;
  if (n >= N) return;
  const float* row = hjk + (size_t)n * 128;
  float r[5];
#pragma unroll
  for (int j = 0; j < 5; ++j) r[j] = sb[j];
  for (int k = 0; k < 128; ++k) {
    float v = row[k];
#pragma unroll
    for (int j = 0; j < 5; ++j) r[j] += v * sw[j * 128 + k];
  }
  float m = r[0];
#pragma unroll
  for (int j = 1; j < 5; ++j) m = fmaxf(m, r[j]);
  float s = 0.f;
#pragma unroll
  for (int j = 0; j < 5; ++j) s += expf(r[j] - m);
  float ls = logf(s);
#pragma unroll
  for (int j = 0; j < 5; ++j) out[(size_t)n * 5 + j] = r[j] - m - ls;
}

// ---------------- host launch ----------------
extern "C" void kernel_launch(void* const* d_in, const int* in_sizes, int n_in,
                              void* d_out, int out_size, void* d_ws, size_t ws_size,
                              hipStream_t stream) {
  (void)n_in; (void)out_size; (void)ws_size;
  const float* x      = (const float*)d_in[0];
  const int*   ei     = (const int*)  d_in[1];
  const float* eattr  = (const float*)d_in[2];
  const float* ee_w1  = (const float*)d_in[3];
  const float* ee_b1  = (const float*)d_in[4];
  const float* ee_w2  = (const float*)d_in[5];
  const float* ee_b2  = (const float*)d_in[6];
  const float* proj_w = (const float*)d_in[7];
  const float* proj_b = (const float*)d_in[8];
  const float* c1_lin = (const float*)d_in[9];
  const float* c1_as  = (const float*)d_in[10];
  const float* c1_ad  = (const float*)d_in[11];
  const float* c1_le  = (const float*)d_in[12];
  const float* c1_ae  = (const float*)d_in[13];
  const float* c1_b   = (const float*)d_in[14];
  const float* c2_lin = (const float*)d_in[15];
  const float* c2_as  = (const float*)d_in[16];
  const float* c2_ad  = (const float*)d_in[17];
  const float* c2_le  = (const float*)d_in[18];
  const float* c2_ae  = (const float*)d_in[19];
  const float* c2_b   = (const float*)d_in[20];
  const float* n1g    = (const float*)d_in[21];
  const float* n1b    = (const float*)d_in[22];
  const float* n2g    = (const float*)d_in[23];
  const float* n2b    = (const float*)d_in[24];
  const float* jkw    = (const float*)d_in[25];
  const float* jkb    = (const float*)d_in[26];
  const float* clsw   = (const float*)d_in[27];
  const float* clsb   = (const float*)d_in[28];

  const int N = in_sizes[0] / 16;      // 50000
  const int E = in_sizes[2] / 8;       // 800000
  const int* src = ei;
  const int* dst = ei + E;

  float* ws = (float*)d_ws;
  size_t off = 0;
  float* ae1  = ws + off; off += (size_t)E * 4;
  float* ae2  = ws + off; off += (size_t)E * 4;
  float* exb  = ws + off; off += (size_t)E * 4;
  float* h0   = ws + off; off += (size_t)N * 128;
  float* xs   = ws + off; off += (size_t)N * 128;   // xs1, then reused as xs2
  float* h1   = ws + off; off += (size_t)N * 128;
  float* agg  = ws + off; off += (size_t)N * 128;   // conv accumulator, then JK output
  float* asrc = ws + off; off += (size_t)N * 4;
  float* adst = ws + off; off += (size_t)N * 4;
  float* amax = ws + off; off += (size_t)N * 4;
  float* den  = ws + off; off += (size_t)N * 4;
  float* Mbuf = ws + off; off += 32;
  half_t* cat = (half_t*)(ws + off); off += (size_t)N * 192;  // [N,384] f16: h0|h1|h2
  half_t* c2h = (half_t*)(ws + off); off += 8192;             // 128x128 f16
  half_t* jkh = (half_t*)(ws + off); off += 24576;            // 128x384 f16

  const int gE256 = (E + 255) / 256;

  k_edge_att_mats<<<1, 32, 0, stream>>>(c1_le, c1_ae, c2_le, c2_ae, Mbuf);
  k_whalf<<<(49152 + 255) / 256, 256, 0, stream>>>(c2_lin, jkw, c2h, jkh);
  k_edge_encoder<<<(E + 127) / 128, 128, 0, stream>>>(eattr, ee_w1, ee_b1, ee_w2, ee_b2,
                                                      Mbuf, ae1, ae2, E);
  k_node_in<<<N, 128, 0, stream>>>(x, proj_w, proj_b, c1_lin, h0, xs, N);

  // ---- conv1 ----
  k_att<<<(N * 4 + 255) / 256, 256, 0, stream>>>(xs, c1_as, c1_ad, asrc, adst, N);
  k_init<<<(N * 128 + 255) / 256, 256, 0, stream>>>(amax, den, agg, N);
  k_edge_alpha<<<gE256, 256, 0, stream>>>(src, dst, asrc, adst, ae1, exb, amax, E);
  k_edge_exp<<<gE256, 256, 0, stream>>>(dst, exb, amax, den, E);
  k_message<<<(E + 15) / 16, 128, 0, stream>>>(src, dst, exb, den, xs, agg, E);
  k_post<<<(N + 7) / 8, 256, 0, stream>>>(agg, c1_b, h0, n1g, n1b, h1, cat, 128, 0, N);

  // ---- conv2 ----
  k_gemm_wmma<128><<<N / 16, 256, 0, stream>>>(cat + 128, 384, c2h, 128, xs, nullptr);
  k_att<<<(N * 4 + 255) / 256, 256, 0, stream>>>(xs, c2_as, c2_ad, asrc, adst, N);
  k_init<<<(N * 128 + 255) / 256, 256, 0, stream>>>(amax, den, agg, N);
  k_edge_alpha<<<gE256, 256, 0, stream>>>(src, dst, asrc, adst, ae2, exb, amax, E);
  k_edge_exp<<<gE256, 256, 0, stream>>>(dst, exb, amax, den, E);
  k_message<<<(E + 15) / 16, 128, 0, stream>>>(src, dst, exb, den, xs, agg, E);
  k_post<<<(N + 7) / 8, 256, 0, stream>>>(agg, c2_b, h1, n2g, n2b, nullptr, cat, 256, -1, N);

  // ---- JK projection (K=384) + classifier ----
  k_gemm_wmma<384><<<N / 16, 256, 0, stream>>>(cat, 384, jkh, 384, agg, jkb);
  k_classifier<<<(N + 255) / 256, 256, 0, stream>>>(agg, clsw, clsb, (float*)d_out, N);
}
